// VLMSDeformAttnLayer_24240795419420
// MI455X (gfx1250) — matrix operand order repaired and checked by reading, not compiled, not run on hardware
//
#include <hip/hip_runtime.h>
#include <hip/hip_bf16.h>
#include <math.h>

// ---------------------------------------------------------------------------
// VLMSDeformAttnLayer for MI455X (gfx1250, wave32, WMMA).
// Dense projections/FFNs -> v_wmma_f32_16x16x32_bf16 (f32 accumulate).
// Big GEMMs: 256Mx64N block tile, LDS-staged bf16 B tile,
//            8 WMMA per wave per 32-K step (2 A frags x 4 B frags).
// p2l attention: split-K online-softmax (partial + combine) for parallelism.
// ---------------------------------------------------------------------------

#define HEADS   8
#define EDIM    256
#define INNER   1024
#define HD      128          // INNER / HEADS
#define NBATCH  2
#define SLEN    40
#define LTOT    21760        // 128^2 + 64^2 + 32^2 + 16^2
#define LN_EPS  1e-5f
#define CLAMPV  50000.0f
#define PREC    132          // partial record: 128 acc + m + s, padded to x4

typedef __attribute__((ext_vector_type(16))) __bf16 v16bf;
typedef __attribute__((ext_vector_type(8)))  float  v8f;

static __device__ __forceinline__ float wave_sum(float v) {
#pragma unroll
  for (int off = 16; off > 0; off >>= 1) v += __shfl_xor(v, off, 32);
  return v;
}

// Build one 16x32 bf16 A fragment for row band starting at row `am`
// (per-lane: one contiguous 16-float run, k = 16*(lane/16) + s).
template<bool HAS_A2>
static __device__ __forceinline__ v16bf load_a_frag(const float* __restrict__ arow,
                                                    const float* __restrict__ arow2,
                                                    int koff) {
  float av[16];
  const float4 a0 = *(const float4*)(arow + koff);
  const float4 a1 = *(const float4*)(arow + koff + 4);
  const float4 a2 = *(const float4*)(arow + koff + 8);
  const float4 a3 = *(const float4*)(arow + koff + 12);
  av[0]=a0.x; av[1]=a0.y; av[2]=a0.z;  av[3]=a0.w;
  av[4]=a1.x; av[5]=a1.y; av[6]=a1.z;  av[7]=a1.w;
  av[8]=a2.x; av[9]=a2.y; av[10]=a2.z; av[11]=a2.w;
  av[12]=a3.x; av[13]=a3.y; av[14]=a3.z; av[15]=a3.w;
  if (HAS_A2) {
    const float4 b0 = *(const float4*)(arow2 + koff);
    const float4 b1 = *(const float4*)(arow2 + koff + 4);
    const float4 b2 = *(const float4*)(arow2 + koff + 8);
    const float4 b3 = *(const float4*)(arow2 + koff + 12);
    av[0]+=b0.x; av[1]+=b0.y; av[2]+=b0.z;  av[3]+=b0.w;
    av[4]+=b1.x; av[5]+=b1.y; av[6]+=b1.z;  av[7]+=b1.w;
    av[8]+=b2.x; av[9]+=b2.y; av[10]+=b2.z; av[11]+=b2.w;
    av[12]+=b3.x; av[13]+=b3.y; av[14]+=b3.z; av[15]+=b3.w;
  }
  v16bf af;
#pragma unroll
  for (int e = 0; e < 16; ++e) af[e] = (__bf16)av[e];
  return af;
}

// ---------------------------------------------------------------------------
// Big GEMM: C[M,Ncol] = epi((A (+A2)) @ B + bias (+resid)),
// M % 256 == 0, Ncol % 64 == 0, K % 32 == 0.
// Block = 256 thr = 8 waves; block tile 256Mx64N; wave tile 32Mx64N:
// 2 A frags reused across 4 LDS B frags -> 8 WMMA per wave per K-step.
// B tile staged in LDS as bf16, [col][k] so a fragment is 32 contiguous bytes.
// ---------------------------------------------------------------------------
template<bool HAS_A2, bool HAS_RES, bool RELU>
__global__ __launch_bounds__(256)
void gemm_big_kernel(const float* __restrict__ A, const float* __restrict__ A2,
                     const float* __restrict__ B, const float* __restrict__ bias,
                     const float* __restrict__ resid, float* __restrict__ C,
                     int M, int K, int Ncol)
{
  __shared__ __align__(32) __bf16 Bs[64 * 32];   // [n_local 0..63][k 0..31]

  const int tid  = threadIdx.x;
  const int lane = tid & 31;
  const int wave = tid >> 5;                      // 0..7 -> 32-row M sub-tile
  const int nblk = blockIdx.x << 6;               // N base
  const int m0   = (blockIdx.y << 8) + (wave << 5);

  // B staging: thread -> (k pair 2*kp, 4 consecutive cols ncl..ncl+3)
  const int kp  = tid >> 4;                       // 0..15
  const int ncl = (tid & 15) << 2;                // 0,4,..,60

  const int am0 = m0 + (lane & 15);               // A row, fragment 0
  const int am1 = am0 + 16;                       // A row, fragment 1
  const float* arow0  = A + (size_t)am0 * K;
  const float* arow1  = A + (size_t)am1 * K;
  const float* arow0b = HAS_A2 ? (A2 + (size_t)am0 * K) : nullptr;
  const float* arow1b = HAS_A2 ? (A2 + (size_t)am1 * K) : nullptr;

  v8f acc[2][4] = {{{}, {}, {}, {}}, {{}, {}, {}, {}}};

  for (int k0 = 0; k0 < K; k0 += 32) {
    // ---- stage B(32x64) -> LDS bf16, coalesced float4 global loads
    {
      const int kk = k0 + 2 * kp;
      const float4 r0 = *(const float4*)(B + (size_t)kk * Ncol + nblk + ncl);
      const float4 r1 = *(const float4*)(B + (size_t)(kk + 1) * Ncol + nblk + ncl);
      const float x0[4] = {r0.x, r0.y, r0.z, r0.w};
      const float x1[4] = {r1.x, r1.y, r1.z, r1.w};
#pragma unroll
      for (int i = 0; i < 4; ++i) {
        union { unsigned u; __bf16 h[2]; } cv;
        cv.h[0] = (__bf16)x0[i];                  // k   (low 16)
        cv.h[1] = (__bf16)x1[i];                  // k+1 (high 16)
        *(unsigned*)&Bs[(ncl + i) * 32 + 2 * kp] = cv.u;
      }
    }
    __syncthreads();

    // ---- 2 A fragments (rows m0..m0+15 and m0+16..m0+31)
    const int koff = k0 + ((lane >> 4) << 4);
    const v16bf af0 = load_a_frag<HAS_A2>(arow0, arow0b, koff);
    const v16bf af1 = load_a_frag<HAS_A2>(arow1, arow1b, koff);

    // ---- 4 B fragments from LDS (each = 32 contiguous bytes) + 8 WMMA
    const int kh = (lane >> 4) << 4;              // 0 or 16
#pragma unroll
    for (int j = 0; j < 4; ++j) {
      const v16bf bfr = *(const v16bf*)&Bs[(j * 16 + (lane & 15)) * 32 + kh];
      acc[0][j] = __builtin_amdgcn_wmma_f32_16x16x32_bf16(
          false, af0, false, bfr, (short)0, acc[0][j], false, false);
      acc[1][j] = __builtin_amdgcn_wmma_f32_16x16x32_bf16(
          false, af1, false, bfr, (short)0, acc[1][j], false, false);
    }
    __syncthreads();
  }

  // ---- epilogue: frag f, VGPR r -> M = m0 + f*16 + (lane>=16 ? 8:0) + r
#pragma unroll
  for (int f = 0; f < 2; ++f) {
    const int mb = m0 + f * 16 + ((lane >> 4) << 3);
#pragma unroll
    for (int j = 0; j < 4; ++j) {
      const int cn = nblk + j * 16 + (lane & 15);
      const float bv = bias[cn];
#pragma unroll
      for (int r = 0; r < 8; ++r) {
        float v = acc[f][j][r] + bv;
        if (HAS_RES) v += resid[(size_t)(mb + r) * Ncol + cn];
        if (RELU)    v = fmaxf(v, 0.0f);
        C[(size_t)(mb + r) * Ncol + cn] = v;
      }
    }
  }
}

// ---------------------------------------------------------------------------
// Small GEMM (lang path, M=80): wave per 16x16 tile, direct global loads.
// ---------------------------------------------------------------------------
template<bool RELU>
__global__ __launch_bounds__(128)
void gemm_small_kernel(const float* __restrict__ A, const float* __restrict__ B,
                       const float* __restrict__ bias, float* __restrict__ C,
                       int M, int K, int Ncol)
{
  const int lane = threadIdx.x & 31;
  const int wave = threadIdx.x >> 5;
  const int n0 = (blockIdx.x << 6) + (wave << 4);
  const int m0 = blockIdx.y << 4;
  if (n0 >= Ncol || m0 >= M) return;

  v8f acc = {};
  const int am = m0 + (lane & 15);
  const int bn = n0 + (lane & 15);
  const float* arow = A + (size_t)am * K;

  for (int k0 = 0; k0 < K; k0 += 32) {
    const int koff = k0 + ((lane >> 4) << 4);
    const v16bf af = load_a_frag<false>(arow, nullptr, koff);
    const int kh = k0 + ((lane >> 4) << 4);
    v16bf bfr;
#pragma unroll
    for (int r = 0; r < 8; ++r) {
      bfr[2 * r]     = (__bf16)B[(size_t)(kh + 2 * r)     * Ncol + bn];
      bfr[2 * r + 1] = (__bf16)B[(size_t)(kh + 2 * r + 1) * Ncol + bn];
    }
    acc = __builtin_amdgcn_wmma_f32_16x16x32_bf16(
        false, af, false, bfr, (short)0, acc, false, false);
  }
  const int mb = m0 + ((lane >> 4) << 3);
  const float bv = bias[bn];
#pragma unroll
  for (int r = 0; r < 8; ++r) {
    float v = acc[r] + bv;
    if (RELU) v = fmaxf(v, 0.0f);
    C[(size_t)(mb + r) * Ncol + bn] = v;
  }
}

// ---------------------------------------------------------------------------
// Single-pass MHA (small Lk): wave per (n,h,q), online softmax.
// ---------------------------------------------------------------------------
__global__ void attn_kernel(const float* __restrict__ Q, const float* __restrict__ Km,
                            const float* __restrict__ V, const int* __restrict__ mask,
                            int invert, float* __restrict__ O,
                            int nb, int Lq, int Lk, float scale)
{
  const int lane = threadIdx.x & 31;
  const int wid = blockIdx.x * (blockDim.x >> 5) + (threadIdx.x >> 5);
  if (wid >= nb * HEADS * Lq) return;
  const int q = wid % Lq;
  const int h = (wid / Lq) % HEADS;
  const int n = wid / (Lq * HEADS);

  float4 qv = *(const float4*)(Q + ((size_t)(n * Lq + q) * INNER + h * HD + lane * 4));
  qv.x *= scale; qv.y *= scale; qv.z *= scale; qv.w *= scale;

  float mrun = -INFINITY, srun = 0.0f;
  float4 acc = make_float4(0.f, 0.f, 0.f, 0.f);
  const int* mrow = mask + (size_t)n * Lk;

  for (int k = 0; k < Lk; ++k) {
    const size_t kbase = (size_t)(n * Lk + k) * INNER + h * HD + lane * 4;
    const float4 kv = *(const float4*)(Km + kbase);
    float logit = wave_sum(qv.x * kv.x + qv.y * kv.y + qv.z * kv.z + qv.w * kv.w);
    logit = fminf(fmaxf(logit, -CLAMPV), CLAMPV);
    const int mv = mrow[k];
    if (!(invert ? (mv == 0) : (mv != 0))) logit = -1.0e9f;
    const float mnew = fmaxf(mrun, logit);
    const float corr = __expf(mrun - mnew);
    const float p    = __expf(logit - mnew);
    const float4 vv = *(const float4*)(V + kbase);
    srun  = srun * corr + p;
    acc.x = acc.x * corr + p * vv.x;  acc.y = acc.y * corr + p * vv.y;
    acc.z = acc.z * corr + p * vv.z;  acc.w = acc.w * corr + p * vv.w;
    mrun = mnew;
  }
  const float inv = 1.0f / srun;
  *(float4*)(O + ((size_t)(n * Lq + q) * INNER + h * HD + lane * 4)) =
      make_float4(acc.x * inv, acc.y * inv, acc.z * inv, acc.w * inv);
}

// ---------------------------------------------------------------------------
// Split-K attention for big Lk (p2l): wave per (n,h,q,chunk).
// Record: [0..127]=unnormalized acc, [128]=chunk max m, [129]=chunk sum s.
// ---------------------------------------------------------------------------
__global__ void attn_partial_kernel(const float* __restrict__ Q,
                                    const float* __restrict__ Km,
                                    const float* __restrict__ V,
                                    const int* __restrict__ mask, int invert,
                                    float* __restrict__ part,
                                    int nb, int Lq, int Lk, int nchunk, int cklen,
                                    float scale)
{
  const int lane = threadIdx.x & 31;
  const int wid = blockIdx.x * (blockDim.x >> 5) + (threadIdx.x >> 5);
  if (wid >= nb * HEADS * Lq * nchunk) return;
  const int c = wid % nchunk;
  const int q = (wid / nchunk) % Lq;
  const int h = (wid / (nchunk * Lq)) % HEADS;
  const int n = wid / (nchunk * Lq * HEADS);

  float4 qv = *(const float4*)(Q + ((size_t)(n * Lq + q) * INNER + h * HD + lane * 4));
  qv.x *= scale; qv.y *= scale; qv.z *= scale; qv.w *= scale;

  const int kbeg = c * cklen;
  const int kend = (kbeg + cklen < Lk) ? (kbeg + cklen) : Lk;
  float mrun = -INFINITY, srun = 0.0f;
  float4 acc = make_float4(0.f, 0.f, 0.f, 0.f);
  const int* mrow = mask + (size_t)n * Lk;

  for (int k = kbeg; k < kend; ++k) {
    const size_t kbase = (size_t)(n * Lk + k) * INNER + h * HD + lane * 4;
    const float4 kv = *(const float4*)(Km + kbase);
    float logit = wave_sum(qv.x * kv.x + qv.y * kv.y + qv.z * kv.z + qv.w * kv.w);
    logit = fminf(fmaxf(logit, -CLAMPV), CLAMPV);
    const int mv = mrow[k];
    if (!(invert ? (mv == 0) : (mv != 0))) logit = -1.0e9f;
    const float mnew = fmaxf(mrun, logit);
    const float corr = __expf(mrun - mnew);
    const float p    = __expf(logit - mnew);
    const float4 vv = *(const float4*)(V + kbase);
    srun  = srun * corr + p;
    acc.x = acc.x * corr + p * vv.x;  acc.y = acc.y * corr + p * vv.y;
    acc.z = acc.z * corr + p * vv.z;  acc.w = acc.w * corr + p * vv.w;
    mrun = mnew;
  }
  const size_t rec = ((((size_t)n * HEADS + h) * Lq + q) * nchunk + c) * PREC;
  *(float4*)(part + rec + lane * 4) = acc;
  if (lane == 0) { part[rec + HD] = mrun; part[rec + HD + 1] = srun; }
}

__global__ void attn_combine_kernel(const float* __restrict__ part,
                                    float* __restrict__ O,
                                    int nb, int Lq, int nchunk)
{
  const int lane = threadIdx.x & 31;
  const int wid = blockIdx.x * (blockDim.x >> 5) + (threadIdx.x >> 5);
  if (wid >= nb * HEADS * Lq) return;
  const int q = wid % Lq;
  const int h = (wid / Lq) % HEADS;
  const int n = wid / (Lq * HEADS);
  const size_t base = (((size_t)n * HEADS + h) * Lq + q) * nchunk;

  float gm = -INFINITY;
  for (int c = 0; c < nchunk; ++c)
    gm = fmaxf(gm, part[(base + c) * PREC + HD]);
  float s = 0.f;
  float4 acc = make_float4(0.f, 0.f, 0.f, 0.f);
  for (int c = 0; c < nchunk; ++c) {
    const size_t rec = (base + c) * PREC;
    const float w = __expf(part[rec + HD] - gm);
    s += w * part[rec + HD + 1];
    const float4 a = *(const float4*)(part + rec + lane * 4);
    acc.x += w * a.x; acc.y += w * a.y; acc.z += w * a.z; acc.w += w * a.w;
  }
  const float inv = 1.0f / s;
  *(float4*)(O + ((size_t)(n * Lq + q) * INNER + h * HD + lane * 4)) =
      make_float4(acc.x * inv, acc.y * inv, acc.z * inv, acc.w * inv);
}

// ---------------------------------------------------------------------------
// LayerNorm over E=256: Y = LN(X (+R)) * g + b. Wave per row; in-place safe.
// ---------------------------------------------------------------------------
__global__ void ln_kernel(const float* __restrict__ X, const float* __restrict__ R,
                          const float* __restrict__ g, const float* __restrict__ b,
                          float* __restrict__ Y, int rows)
{
  const int lane = threadIdx.x & 31;
  const int row = blockIdx.x * (blockDim.x >> 5) + (threadIdx.x >> 5);
  if (row >= rows) return;
  const float* xr = X + (size_t)row * EDIM;
  const float* rr = R ? (R + (size_t)row * EDIM) : nullptr;

  float v[8];
  float s = 0.f;
#pragma unroll
  for (int i = 0; i < 8; ++i) {
    const int c = i * 32 + lane;
    float x = xr[c];
    if (rr) x += rr[c];
    v[i] = x; s += x;
  }
  const float mean = wave_sum(s) * (1.0f / EDIM);
  float var = 0.f;
#pragma unroll
  for (int i = 0; i < 8; ++i) { const float d = v[i] - mean; var += d * d; }
  var = wave_sum(var) * (1.0f / EDIM);
  const float rs = rsqrtf(var + LN_EPS);
#pragma unroll
  for (int i = 0; i < 8; ++i) {
    const int c = i * 32 + lane;
    Y[(size_t)row * EDIM + c] = (v[i] - mean) * rs * g[c] + b[c];
  }
}

// ---------------------------------------------------------------------------
// Deformable sampling: wave per (n,q,head); lane = channel within head (32).
// ---------------------------------------------------------------------------
__global__ void deform_sample_kernel(const float* __restrict__ val,
                                     const float* __restrict__ off,
                                     const float* __restrict__ aw,
                                     const float* __restrict__ refp,
                                     const int* __restrict__ kpm,
                                     float* __restrict__ out)
{
  const int sh[4] = {128, 64, 32, 16};
  const int sw[4] = {128, 64, 32, 16};
  const int st[4] = {0, 16384, 20480, 21504};

  const int lane = threadIdx.x & 31;
  const int wid = blockIdx.x * (blockDim.x >> 5) + (threadIdx.x >> 5);
  if (wid >= NBATCH * LTOT * HEADS) return;
  const int h = wid % HEADS;
  const int q = (wid / HEADS) % LTOT;
  const int n = wid / (HEADS * LTOT);
  const size_t row = (size_t)n * LTOT + q;

  float lg[16];
  float mx = -INFINITY;
#pragma unroll
  for (int j = 0; j < 16; ++j) { lg[j] = aw[row * 128 + h * 16 + j]; mx = fmaxf(mx, lg[j]); }
  float ssum = 0.f;
#pragma unroll
  for (int j = 0; j < 16; ++j) { lg[j] = __expf(lg[j] - mx); ssum += lg[j]; }
  const float sinv = 1.0f / ssum;

  float accum = 0.f;
  for (int lvl = 0; lvl < 4; ++lvl) {
    const int hh = sh[lvl], ww = sw[lvl], start = st[lvl];
    const float rx = refp[(row * 4 + lvl) * 2 + 0];
    const float ry = refp[(row * 4 + lvl) * 2 + 1];
#pragma unroll
    for (int pt = 0; pt < 4; ++pt) {
      const int oc = ((h * 4 + lvl) * 4 + pt) * 2;
      const float locx = rx + off[row * 256 + oc]     / (float)ww;   // norm=[w,h]
      const float locy = ry + off[row * 256 + oc + 1] / (float)hh;
      const float x = locx * (float)ww - 0.5f;
      const float y = locy * (float)hh - 0.5f;
      const float x0f = floorf(x), y0f = floorf(y);
      const float lx = x - x0f, ly = y - y0f;
      const int x0 = (int)x0f, y0 = (int)y0f;
      float sample = 0.f;
#pragma unroll
      for (int c = 0; c < 4; ++c) {
        const int dx = c & 1, dy = c >> 1;
        const float w = (dx ? lx : 1.f - lx) * (dy ? ly : 1.f - ly);
        const int xi = x0 + dx, yi = y0 + dy;
        const bool valid = (xi >= 0) && (xi < ww) && (yi >= 0) && (yi < hh);
        const int xc = xi < 0 ? 0 : (xi > ww - 1 ? ww - 1 : xi);
        const int yc = yi < 0 ? 0 : (yi > hh - 1 ? hh - 1 : yi);
        const size_t vrow = (size_t)n * LTOT + start + yc * ww + xc;
        float gv = val[vrow * 256 + h * 32 + lane];
        if (kpm[vrow] != 0) gv = 0.f;
        sample += valid ? (w * gv) : 0.f;
      }
      accum += sample * lg[lvl * 4 + pt] * sinv;
    }
  }
  out[row * 256 + h * 32 + lane] = accum;
}

// ---------------------------------------------------------------------------
// Host-side orchestration
// ---------------------------------------------------------------------------
static inline void gemm_big(hipStream_t s, const float* A, const float* A2,
                            const float* B, const float* bias, const float* resid,
                            float* C, int M, int K, int Ncol, int relu) {
  dim3 grid(Ncol / 64, M / 256);
  if (A2)        gemm_big_kernel<true,  false, false><<<grid, 256, 0, s>>>(A, A2, B, bias, resid, C, M, K, Ncol);
  else if (resid)gemm_big_kernel<false, true,  false><<<grid, 256, 0, s>>>(A, A2, B, bias, resid, C, M, K, Ncol);
  else if (relu) gemm_big_kernel<false, false, true ><<<grid, 256, 0, s>>>(A, A2, B, bias, resid, C, M, K, Ncol);
  else           gemm_big_kernel<false, false, false><<<grid, 256, 0, s>>>(A, A2, B, bias, resid, C, M, K, Ncol);
}
static inline void gemm_small(hipStream_t s, const float* A, const float* B,
                              const float* bias, float* C, int M, int K, int Ncol,
                              int relu) {
  dim3 grid(Ncol / 64, (M + 15) / 16);
  if (relu) gemm_small_kernel<true ><<<grid, 128, 0, s>>>(A, B, bias, C, M, K, Ncol);
  else      gemm_small_kernel<false><<<grid, 128, 0, s>>>(A, B, bias, C, M, K, Ncol);
}
static inline void attn(hipStream_t s, const float* Q, const float* K,
                        const float* V, const int* mask, int invert, float* O,
                        int nb, int Lq, int Lk) {
  const int waves = nb * HEADS * Lq;
  attn_kernel<<<(waves * 32 + 255) / 256, 256, 0, s>>>(
      Q, K, V, mask, invert, O, nb, Lq, Lk, 0.08838834764831845f);
}
static inline void attn_split(hipStream_t s, const float* Q, const float* K,
                              const float* V, const int* mask, int invert,
                              float* part, float* O, int nb, int Lq, int Lk,
                              int nchunk) {
  const int cklen = (Lk + nchunk - 1) / nchunk;
  const int pw = nb * HEADS * Lq * nchunk;
  attn_partial_kernel<<<(pw * 32 + 255) / 256, 256, 0, s>>>(
      Q, K, V, mask, invert, part, nb, Lq, Lk, nchunk, cklen, 0.08838834764831845f);
  const int cw = nb * HEADS * Lq;
  attn_combine_kernel<<<(cw * 32 + 255) / 256, 256, 0, s>>>(part, O, nb, Lq, nchunk);
}
static inline void layernorm(hipStream_t s, const float* X, const float* R,
                             const float* g, const float* b, float* Y, int rows) {
  ln_kernel<<<(rows + 7) / 8, 256, 0, s>>>(X, R, g, b, Y, rows);
}

extern "C" void kernel_launch(void* const* d_in, const int* in_sizes, int n_in,
                              void* d_out, int out_size, void* d_ws, size_t ws_size,
                              hipStream_t stream) {
  (void)in_sizes; (void)n_in; (void)out_size; (void)ws_size;

  const float* vis      = (const float*)d_in[0];   // (N,L,E)
  const float* lang     = (const float*)d_in[1];   // (N,S,E)
  const int*   lang_msk = (const int*)  d_in[2];   // (N,S)
  const float* vis_pos  = (const float*)d_in[3];   // (N,L,E)
  const int*   vis_pad  = (const int*)  d_in[4];   // (N,L)
  const float* refp     = (const float*)d_in[5];   // (N,L,4,2)
#define PW(i) ((const float*)d_in[(i)])
  // params depth-first from index 8:
  // p2l{q,k,v,o}{w,b}=8..15  lself=16..23  l2p=24..31
  // lmlp=32..35  vmlp=36..39  deform{value,off,attw,out}{w,b}=40..47
  // ln1..ln3,vn1..vn3{g,b}=48..59

  const size_t NL = (size_t)NBATCH * LTOT;   // 43520
  const size_t NS = (size_t)NBATCH * SLEN;   // 80
  float* W    = (float*)d_ws;
  float* bufA = W;                            // NL*1024
  float* bufB = bufA + NL * 1024;             // NL*1024
  float* buf1 = bufB + NL * 1024;             // NL*256  (pv)
  float* buf2 = buf1 + NL * 256;              // NL*256
  float* buf3 = buf2 + NL * 256;              // NL*256
  float* buf4 = buf3 + NL * 256;              // NL*128
  float* buf5 = buf4 + NL * 128;              // NL*256
  float* buf6 = buf5 + NL * 256;              // NL*256
  float* lq   = buf6 + NL * 256;              // NS*1024 each
  float* lk   = lq  + NS * 1024;
  float* lv   = lk  + NS * 1024;
  float* lo   = lv  + NS * 1024;
  float* lt1  = lo  + NS * 1024;
  float* lt2  = lt1 + NS * 1024;
  float* lt3  = lt2 + NS * 1024;
  float* part = lt3 + NS * 1024;              // 2*8*40*20*PREC floats (~6.8MB)
  const int NCHUNK = 20;

  float* out_v    = (float*)d_out;            // (N,L,E)
  float* out_lang = out_v + NL * EDIM;        // (N,S,E)

  const int ML = (int)NL, MS = (int)NS;

  // ===== Lang branch =====
  gemm_big(stream, vis, vis_pos, PW(10), PW(11), nullptr, bufA, ML, 256, 1024, 0); // kh
  gemm_big(stream, vis, nullptr, PW(12), PW(13), nullptr, bufB, ML, 256, 1024, 0); // vh
  gemm_small(stream, lang, PW(8), PW(9), lq, MS, 256, 1024, 0);                    // qh
  attn_split(stream, lq, bufA, bufB, vis_pad, /*invert=*/1, part, lo,
             NBATCH, SLEN, LTOT, NCHUNK);
  gemm_small(stream, lo, PW(14), PW(15), lt1, MS, 1024, 256, 0);                   // o-proj
  layernorm(stream, lt1, lang, PW(48), PW(49), lt2, MS);                           // ln1

  gemm_small(stream, lt2, PW(16), PW(17), lq, MS, 256, 1024, 0);
  gemm_small(stream, lt2, PW(18), PW(19), lk, MS, 256, 1024, 0);
  gemm_small(stream, lt2, PW(20), PW(21), lv, MS, 256, 1024, 0);
  attn(stream, lq, lk, lv, lang_msk, 0, lo, NBATCH, SLEN, SLEN);
  gemm_small(stream, lo, PW(22), PW(23), lt1, MS, 1024, 256, 0);
  layernorm(stream, lt1, lt2, PW(50), PW(51), lt3, MS);                            // ln2

  gemm_small(stream, lt3, PW(32), PW(33), lq, MS, 256, 1024, 1);                   // relu
  gemm_small(stream, lq, PW(34), PW(35), lt1, MS, 1024, 256, 0);
  layernorm(stream, lt1, lt3, PW(52), PW(53), out_lang, MS);                       // ln3

  // ===== Vis branch =====
  gemm_big(stream, vis, vis_pos, PW(24), PW(25), nullptr, bufA, ML, 256, 1024, 0); // q
  gemm_small(stream, out_lang, PW(26), PW(27), lk, MS, 256, 1024, 0);
  gemm_small(stream, out_lang, PW(28), PW(29), lv, MS, 256, 1024, 0);
  attn(stream, bufA, lk, lv, lang_msk, 0, bufB, NBATCH, LTOT, SLEN);
  gemm_big(stream, bufB, nullptr, PW(30), PW(31), vis, buf1, ML, 1024, 256, 0);    // +vis
  layernorm(stream, buf1, nullptr, PW(56), PW(57), buf1, ML);                      // vn2

  gemm_big(stream, buf1, nullptr, PW(40), PW(41), nullptr, buf2, ML, 256, 256, 0); // value
  gemm_big(stream, buf1, vis_pos, PW(42), PW(43), nullptr, buf3, ML, 256, 256, 0); // off
  gemm_big(stream, buf1, vis_pos, PW(44), PW(45), nullptr, buf4, ML, 256, 128, 0); // attw
  {
    const int waves = NBATCH * LTOT * HEADS;
    deform_sample_kernel<<<(waves * 32 + 255) / 256, 256, 0, stream>>>(
        buf2, buf3, buf4, refp, vis_pad, buf5);
  }
  gemm_big(stream, buf5, nullptr, PW(46), PW(47), buf1, buf6, ML, 256, 256, 0);    // +identity
  layernorm(stream, buf6, nullptr, PW(54), PW(55), buf6, ML);                      // vn1

  gemm_big(stream, buf6, nullptr, PW(36), PW(37), nullptr, bufA, ML, 256, 1024, 1);// relu
  gemm_big(stream, bufA, nullptr, PW(38), PW(39), nullptr, buf2, ML, 1024, 256, 0);
  layernorm(stream, buf2, buf6, PW(58), PW(59), out_v, ML);                        // vn3
#undef PW
}
